// QuantumHamiltonian_76089640616806
// MI455X (gfx1250) — matrix operand order repaired and checked by reading, not compiled, not run on hardware
//
#include <hip/hip_runtime.h>

// ---------------------------------------------------------------------------
// QuantumHamiltonian for MI455X (gfx1250, wave32, WMMA).
//
// out[b,t,c,:] = psi[b,t,c,:] @ Re(expm(H1 - i*H0))      (psi is real)
//
// Stage 1: U = expm(M), M = H1 - i*H0, via scaling-and-squaring (s=2) +
//          order-10 Taylor (Horner). 11 sequential complex 256x256 matmuls
//          using V_WMMA_F32_16X16X4_F32 (full fp32 precision).
// Stage 2: 131072x256x256 GEMM with V_WMMA_F32_16X16X32_BF16, Re(U) cached
//          in LDS (bf16, transposed, padded), psi streamed with b128 loads.
// ---------------------------------------------------------------------------

typedef __attribute__((ext_vector_type(16))) __bf16       v16bf;
typedef __attribute__((ext_vector_type(8)))  float        v8f;
typedef __attribute__((ext_vector_type(4)))  float        v4f;
typedef __attribute__((ext_vector_type(2)))  float        v2f;
typedef __attribute__((ext_vector_type(4)))  unsigned int v4u;

#define DIM   256
#define NMAT  (DIM * DIM)          // 65536 elements per 256x256 plane

// ---------------------------------------------------------------------------
// Stage-1 setup: A = (H1 - i*H0) / 2^s  (s = 2)   and   R = I + A/N  (N = 10)
// ---------------------------------------------------------------------------
__global__ __launch_bounds__(256) void expm_setup(const float* __restrict__ H,
                                                  float* __restrict__ Ar,
                                                  float* __restrict__ Ai,
                                                  float* __restrict__ Rr,
                                                  float* __restrict__ Ri) {
  int i = blockIdx.x * 256 + threadIdx.x;     // 0 .. 65535
  float h0 = H[i];                            // H[0] plane
  float h1 = H[NMAT + i];                     // H[1] plane
  float ar = h1 * 0.25f;                      // / 2^2
  float ai = -h0 * 0.25f;
  Ar[i] = ar;
  Ai[i] = ai;
  int r = i >> 8, c = i & 255;
  float d = (r == c) ? 1.0f : 0.0f;
  Rr[i] = d + ar * 0.1f;                      // I + A/10
  Ri[i] = ai * 0.1f;
}

// ---------------------------------------------------------------------------
// Complex 256x256 matmul step:  O = addI*I + alpha * (X @ Y)     (complex)
// One wave computes one 16x16 complex output tile using fp32 WMMA.
// Cr = sum XrYr - sum XiYi ; Ci = sum XrYi + sum XiYr  (4 accumulators,
// since F32 WMMA has no A/B negation modifier).
// Grid: 256 tiles, 4 waves (128 threads) per block -> 64 blocks.
// ---------------------------------------------------------------------------
__global__ __launch_bounds__(128) void cmm256(const float* __restrict__ Xr,
                                              const float* __restrict__ Xi,
                                              const float* __restrict__ Yr,
                                              const float* __restrict__ Yi,
                                              float* __restrict__ Or,
                                              float* __restrict__ Oi,
                                              float alpha, int addI) {
  int tid   = threadIdx.x;
  int wave  = tid >> 5;
  int lane  = tid & 31;
  int lhalf = lane >> 4;          // 0: lanes 0-15, 1: lanes 16-31
  int lrow  = lane & 15;

  int tile = blockIdx.x * 4 + wave;   // 0..255
  int tm   = tile >> 4;
  int tn   = tile & 15;

  int arow = tm * 16 + lrow;          // A-matrix row for this lane
  int bcol = tn * 16 + lrow;          // B-matrix col for this lane

  v8f p1 = {}, p2 = {}, p3 = {}, p4 = {};

  // K loop, 4 per WMMA. F32 A 16x4 frag: lanes 0-15 hold K={k0,k0+1},
  // lanes 16-31 hold K={k0+2,k0+3}; B 4x16 frag mirrored.
  for (int k0 = 0; k0 < DIM; k0 += 4) {
    int ka = k0 + lhalf * 2;
    v2f xr = *(const v2f*)(Xr + arow * DIM + ka);
    v2f xi = *(const v2f*)(Xi + arow * DIM + ka);
    v2f yr, yi;
    yr.x = Yr[ka * DIM + bcol];
    yr.y = Yr[(ka + 1) * DIM + bcol];
    yi.x = Yi[ka * DIM + bcol];
    yi.y = Yi[(ka + 1) * DIM + bcol];
    p1 = __builtin_amdgcn_wmma_f32_16x16x4_f32(false, xr, false, yr, (short)0, p1, false, false);
    p2 = __builtin_amdgcn_wmma_f32_16x16x4_f32(false, xi, false, yi, (short)0, p2, false, false);
    p3 = __builtin_amdgcn_wmma_f32_16x16x4_f32(false, xr, false, yi, (short)0, p3, false, false);
    p4 = __builtin_amdgcn_wmma_f32_16x16x4_f32(false, xi, false, yr, (short)0, p4, false, false);
  }

  // C/D layout: VGPR j -> row tm*16 + j + 8*lhalf, col tn*16 + lrow
#pragma unroll
  for (int j = 0; j < 8; ++j) {
    int r = tm * 16 + lhalf * 8 + j;
    int c = tn * 16 + lrow;
    float vr = alpha * (p1[j] - p2[j]);
    float vi = alpha * (p3[j] + p4[j]);
    if (addI && (r == c)) vr += 1.0f;
    Or[r * DIM + c] = vr;
    Oi[r * DIM + c] = vi;
  }
}

// ---------------------------------------------------------------------------
// Stage 2: out[M=131072, 256] = psi @ Ur, bf16 WMMA 16x16x32, f32 accum.
// Block = 256 threads (8 waves); each wave owns a 16-row strip x all 256 cols
// (16 accumulator tiles). Ur lives in LDS transposed [e][d] as bf16 with
// padded K-stride 264 (528B = 33*16B: b128-aligned, banks spread).
// ---------------------------------------------------------------------------
#define BSTRIDE 264                     // padded K stride in elements

__global__ __launch_bounds__(256) void gemm_psiU(const float* __restrict__ psi,
                                                 const float* __restrict__ Ur,
                                                 float* __restrict__ out) {
  __shared__ __align__(16) unsigned short Bs[DIM * BSTRIDE];  // 132 KB

  int tid = threadIdx.x;

  // Cooperative fill: read Ur row-major (coalesced), store transposed bf16.
  for (int i = tid; i < NMAT; i += 256) {
    int d = i >> 8;            // K index (input dim)
    int e = i & 255;           // N index (output dim)
    unsigned int bits = __float_as_uint(Ur[i]);
    Bs[e * BSTRIDE + d] = (unsigned short)(bits >> 16);   // f32 -> bf16 (trunc)
  }
  __syncthreads();

  int wave  = tid >> 5;
  int lane  = tid & 31;
  int lhalf = lane >> 4;
  int lrow  = lane & 15;

  long m0 = (long)blockIdx.x * 128 + (long)wave * 16;
  const float* aBase = psi + (m0 + lrow) * DIM;

  v8f acc[16] = {};

  for (int kk = 0; kk < 8; ++kk) {
    int k0 = kk * 32;
    int c0 = k0 + lhalf * 8;    // this lane's first K chunk

    // A frag (16x32 bf16): lane holds K=[c0,c0+8) in u[0..3] and
    // K=[c0+16,c0+24) in u[4..7]. Load f32, pack to bf16 in-register.
    v4f f0 = *(const v4f*)(aBase + c0);
    v4f f1 = *(const v4f*)(aBase + c0 + 4);
    v4f f2 = *(const v4f*)(aBase + c0 + 16);
    v4f f3 = *(const v4f*)(aBase + c0 + 20);

    union { v16bf v; unsigned int u[8]; } aF;
    aF.u[0] = (__float_as_uint(f0.x) >> 16) | (__float_as_uint(f0.y) & 0xFFFF0000u);
    aF.u[1] = (__float_as_uint(f0.z) >> 16) | (__float_as_uint(f0.w) & 0xFFFF0000u);
    aF.u[2] = (__float_as_uint(f1.x) >> 16) | (__float_as_uint(f1.y) & 0xFFFF0000u);
    aF.u[3] = (__float_as_uint(f1.z) >> 16) | (__float_as_uint(f1.w) & 0xFFFF0000u);
    aF.u[4] = (__float_as_uint(f2.x) >> 16) | (__float_as_uint(f2.y) & 0xFFFF0000u);
    aF.u[5] = (__float_as_uint(f2.z) >> 16) | (__float_as_uint(f2.w) & 0xFFFF0000u);
    aF.u[6] = (__float_as_uint(f3.x) >> 16) | (__float_as_uint(f3.y) & 0xFFFF0000u);
    aF.u[7] = (__float_as_uint(f3.z) >> 16) | (__float_as_uint(f3.w) & 0xFFFF0000u);

#pragma unroll
    for (int n = 0; n < 16; ++n) {
      // B frag (32x16 bf16): lane = col N (lrow), K matches A:
      // lane holds K=[c0,c0+8) in q[0] and K=[c0+16,c0+24) in q[1].
      const unsigned short* bp = &Bs[(n * 16 + lrow) * BSTRIDE + k0 + lhalf * 8];
      union { v16bf v; v4u q[2]; } bF;
      bF.q[0] = *(const v4u*)(bp);
      bF.q[1] = *(const v4u*)(bp + 16);
      acc[n] = __builtin_amdgcn_wmma_f32_16x16x32_bf16(false, aF.v, false, bF.v,
                                                       (short)0, acc[n], false, false);
    }
  }

  // Store: acc[n][j] -> out[m0 + 8*lhalf + j, n*16 + lrow]
#pragma unroll
  for (int n = 0; n < 16; ++n) {
    float* op = out + (m0 + lhalf * 8) * DIM + n * 16 + lrow;
#pragma unroll
    for (int j = 0; j < 8; ++j) op[j * DIM] = acc[n][j];
  }
}

// ---------------------------------------------------------------------------
// Host-side launch sequence (graph-capture safe: only kernel launches).
// ---------------------------------------------------------------------------
extern "C" void kernel_launch(void* const* d_in, const int* in_sizes, int n_in,
                              void* d_out, int out_size, void* d_ws, size_t ws_size,
                              hipStream_t stream) {
  (void)in_sizes; (void)n_in; (void)out_size; (void)ws_size;

  const float* psi = (const float*)d_in[0];   // [8,4096,4,256] f32
  const float* H   = (const float*)d_in[1];   // [4,256,256]    f32
  float* out = (float*)d_out;                 // [8,4096,4,256] f32
  float* ws  = (float*)d_ws;                  // needs 6 * 256KB = 1.5 MB

  float* Ar  = ws + 0 * NMAT;
  float* Ai  = ws + 1 * NMAT;
  float* b0r = ws + 2 * NMAT;
  float* b0i = ws + 3 * NMAT;
  float* b1r = ws + 4 * NMAT;
  float* b1i = ws + 5 * NMAT;

  // A = (H1 - i H0)/4 ; R = I + A/10
  expm_setup<<<NMAT / 256, 256, 0, stream>>>(H, Ar, Ai, b0r, b0i);

  float* Rr = b0r; float* Ri = b0i;
  float* Sr = b1r; float* Si = b1i;

  // Horner: for k = 9..1:  R <- I + (A/k) * R
  for (int k = 9; k >= 1; --k) {
    cmm256<<<64, 128, 0, stream>>>(Ar, Ai, Rr, Ri, Sr, Si, 1.0f / (float)k, 1);
    float* t;
    t = Rr; Rr = Sr; Sr = t;
    t = Ri; Ri = Si; Si = t;
  }

  // Squaring: R <- R*R, twice  (undo the /4 scaling)
  for (int s = 0; s < 2; ++s) {
    cmm256<<<64, 128, 0, stream>>>(Rr, Ri, Rr, Ri, Sr, Si, 1.0f, 0);
    float* t;
    t = Rr; Rr = Sr; Sr = t;
    t = Ri; Ri = Si; Si = t;
  }

  // out = psi @ Re(U).  131072 rows / 128 per block = 1024 blocks.
  gemm_psiU<<<1024, 256, 0, stream>>>(psi, Rr, out);
}
